// GCN_45664092291083
// MI455X (gfx1250) — matrix operand order repaired and checked by reading, not compiled, not run on hardware
//
#include <hip/hip_runtime.h>
#include <hip/hip_bf16.h>
#include <stdint.h>

// ---------------------------------------------------------------------------
// GCN forward on MI455X (gfx1250, wave32).
//   N=16384 nodes, E=524288 edges, IN_DIM=512, HID=256, NCLS=64.
//   A_hat kept as a 16384x16384 bitmap (32MB, fits in 192MB L2).
//   SpMM = dis-weighted row gathers (memory/L2 bound).
//   Linear layers = v_wmma_f32_16x16x32_bf16 tiles (compute part).
// ---------------------------------------------------------------------------

#define NN      16384
#define EE      524288
#define INDIM   512
#define HID     256
#define NCLS    64
#define WORDS_PER_ROW (NN / 32)   // 512

typedef __attribute__((ext_vector_type(16))) __bf16 v16bf;
typedef __attribute__((ext_vector_type(8)))  float  v8f;

// ---------------- bitmap zero ----------------
__global__ void gcn_zero_kernel(uint4* __restrict__ p, long n16) {
    long i = (long)blockIdx.x * blockDim.x + threadIdx.x;
    if (i < n16) p[i] = uint4{0u, 0u, 0u, 0u};
}

// ---------------- edge scatter: set bits (r,c) and (c,r) ----------------
__global__ void gcn_scatter_kernel(const long long* __restrict__ ei,
                                   unsigned int* __restrict__ bm) {
    int e = blockIdx.x * blockDim.x + threadIdx.x;
    if (e >= EE) return;
    int r = (int)ei[e];
    int c = (int)ei[EE + e];
    atomicOr(&bm[(size_t)r * WORDS_PER_ROW + (c >> 5)], 1u << (c & 31));
    atomicOr(&bm[(size_t)c * WORDS_PER_ROW + (r >> 5)], 1u << (r & 31));
}

// ---------------- degree -> dis = (deg + 1 + 1e-8)^-0.5 ----------------
// One wave32 per row; 8 rows per 256-thread block.
__global__ void gcn_degree_kernel(const unsigned int* __restrict__ bm,
                                  float* __restrict__ dis) {
    int row  = blockIdx.x * 8 + (threadIdx.x >> 5);
    int lane = threadIdx.x & 31;
    int cnt = 0;
    const unsigned int* r = bm + (size_t)row * WORDS_PER_ROW;
    for (int wi = lane; wi < WORDS_PER_ROW; wi += 32)
        cnt += __builtin_popcount(r[wi]);
    #pragma unroll
    for (int off = 16; off > 0; off >>= 1)
        cnt += __shfl_xor(cnt, off, 32);
    if (lane == 0) {
        // +1 for the identity term of A_hat = bin(A+A^T) + I
        dis[row] = rsqrtf((float)(cnt + 1) + 1e-8f);
    }
}

// ---------------- fp32 -> bf16 convert ----------------
__global__ void gcn_f2bf_kernel(const float* __restrict__ in,
                                __bf16* __restrict__ out, int n) {
    int i = blockIdx.x * blockDim.x + threadIdx.x;
    if (i < n) out[i] = (__bf16)in[i];
}

// ---------------- SpMM: H[i,:] = dis_i * (sum_{j in nbr(i)} dis_j*X[j,:] + dis_i*X[i,:])
// One 256-thread block per row; bitmap word + dis[j] are wave-uniform (scalar path).
template <int DIM, typename InT>
__global__ void gcn_spmm_kernel(const unsigned int* __restrict__ bm,
                                const float* __restrict__ dis,
                                const InT* __restrict__ X,
                                __bf16* __restrict__ Hout) {
    const int row = blockIdx.x;
    const int t   = threadIdx.x;
    constexpr int CPT = DIM / 256;
    float acc[CPT];
    const float di = dis[row];
    // identity term (weight 1 in A_raw; self-loop edge bit, if set, adds in the loop)
    #pragma unroll
    for (int cc = 0; cc < CPT; ++cc)
        acc[cc] = di * (float)X[(size_t)row * DIM + t + cc * 256];

    const unsigned int* r = bm + (size_t)row * WORDS_PER_ROW;
    for (int wi = 0; wi < WORDS_PER_ROW; ++wi) {
        unsigned int w = r[wi];            // uniform -> scalar load
        while (w) {
            int b = __builtin_ctz(w);
            w &= w - 1;
            int j = (wi << 5) + b;
            float dj = dis[j];             // uniform -> scalar load
            #pragma unroll
            for (int cc = 0; cc < CPT; ++cc)
                acc[cc] += dj * (float)X[(size_t)j * DIM + t + cc * 256];
        }
    }
    #pragma unroll
    for (int cc = 0; cc < CPT; ++cc)
        Hout[(size_t)row * DIM + t + cc * 256] = (__bf16)(di * acc[cc]);
}

// ---------------- dense Linear via WMMA: out[m,n] = act(sum_k A[m,k]*W[n,k] + b[n])
// One wave32 per 16x16 C tile. A/B per-lane layouts per CDNA5 ISA:
//   A (16x32 bf16): lane L -> row m0+(L%16), K in [8g,8g+8) U [16+8g,24+8g), g=L/16
//   B (32x16 bf16): lane L -> col n0+(L%16) (= row of W), K in [8g,8g+8) U [16+8g,24+8g)
//   C/D (16x16 f32): lane L -> col n0+(L%16), rows m0 + v + 8g for VGPR v=0..7
template <int K, int N, bool RELU, typename OutT>
__global__ void gcn_lin_wmma_kernel(const __bf16* __restrict__ A,
                                    const __bf16* __restrict__ W,
                                    const float* __restrict__ bias,
                                    OutT* __restrict__ out) {
    const int lane   = threadIdx.x & 31;
    const int waveId = blockIdx.x * (blockDim.x >> 5) + (threadIdx.x >> 5);
    constexpr int nTiles = N / 16;
    const int mTile = waveId / nTiles;
    const int nTile = waveId % nTiles;
    const int m0 = mTile * 16, n0 = nTile * 16;
    const int g  = lane >> 4;
    const int mr = m0 + (lane & 15);   // A row this lane carries
    const int nr = n0 + (lane & 15);   // W row (= B column) this lane carries

    union { uint4 q[2]; v16bf v; } ua, ub;
    v8f c = {};
    #pragma unroll
    for (int k0 = 0; k0 < K; k0 += 32) {
        ua.q[0] = *(const uint4*)(A + (size_t)mr * K + k0 + 8 * g);
        ua.q[1] = *(const uint4*)(A + (size_t)mr * K + k0 + 16 + 8 * g);
        ub.q[0] = *(const uint4*)(W + (size_t)nr * K + k0 + 8 * g);
        ub.q[1] = *(const uint4*)(W + (size_t)nr * K + k0 + 16 + 8 * g);
        c = __builtin_amdgcn_wmma_f32_16x16x32_bf16(
                /*neg_a=*/false, ua.v, /*neg_b=*/false, ub.v,
                /*c_mod=*/(short)0, c, /*reuse_a=*/false, /*reuse_b=*/false);
    }
    const float bn = bias[nr];
    #pragma unroll
    for (int v = 0; v < 8; ++v) {
        int m = m0 + v + 8 * g;
        float r = c[v] + bn;
        if (RELU) r = fmaxf(r, 0.0f);
        out[(size_t)m * N + nr] = (OutT)r;
    }
}

// ---------------------------------------------------------------------------
extern "C" void kernel_launch(void* const* d_in, const int* /*in_sizes*/, int /*n_in*/,
                              void* d_out, int /*out_size*/, void* d_ws, size_t /*ws_size*/,
                              hipStream_t stream) {
    const float*     X  = (const float*)d_in[0];
    const long long* EI = (const long long*)d_in[1];
    const float*     W1 = (const float*)d_in[2];
    const float*     b1 = (const float*)d_in[3];
    const float*     W2 = (const float*)d_in[4];
    const float*     b2 = (const float*)d_in[5];
    float*           out = (float*)d_out;

    uint8_t* ws = (uint8_t*)d_ws;
    size_t off = 0;
    unsigned int* bm = (unsigned int*)(ws + off); off += (size_t)NN * WORDS_PER_ROW * 4; // 32 MB
    float*  dis = (float*) (ws + off); off += (size_t)NN * 4;                            // 64 KB
    __bf16* H1  = (__bf16*)(ws + off); off += (size_t)NN * INDIM * 2;                    // 16 MB
    __bf16* H2  = (__bf16*)(ws + off); off += (size_t)NN * HID * 2;                      //  8 MB
    __bf16* H3  = (__bf16*)(ws + off); off += (size_t)NN * HID * 2;                      //  8 MB
    __bf16* W1b = (__bf16*)(ws + off); off += (size_t)HID * INDIM * 2;                   // 256 KB
    __bf16* W2b = (__bf16*)(ws + off); off += (size_t)NCLS * HID * 2;                    //  32 KB

    // 1) adjacency bitmap
    long n16 = (long)NN * WORDS_PER_ROW * 4 / 16;
    gcn_zero_kernel<<<(unsigned)((n16 + 255) / 256), 256, 0, stream>>>((uint4*)bm, n16);
    gcn_scatter_kernel<<<(EE + 255) / 256, 256, 0, stream>>>(EI, bm);
    gcn_degree_kernel<<<NN / 8, 256, 0, stream>>>(bm, dis);

    // 2) weights to bf16
    gcn_f2bf_kernel<<<(HID * INDIM + 255) / 256, 256, 0, stream>>>(W1, W1b, HID * INDIM);
    gcn_f2bf_kernel<<<(NCLS * HID + 255) / 256, 256, 0, stream>>>(W2, W2b, NCLS * HID);

    // 3) H1 = A_hat @ X          (fp32 gather, bf16 store)
    gcn_spmm_kernel<INDIM, float><<<NN, 256, 0, stream>>>(bm, dis, X, H1);

    // 4) H2 = relu(H1 @ W1^T + b1)   [16384x512x256] via WMMA bf16
    gcn_lin_wmma_kernel<INDIM, HID, true, __bf16>
        <<<(NN / 16) * (HID / 16) / 8, 256, 0, stream>>>(H1, W1b, b1, H2);

    // 5) H3 = A_hat @ H2         (bf16 gather)
    gcn_spmm_kernel<HID, __bf16><<<NN, 256, 0, stream>>>(bm, dis, H2, H3);

    // 6) out = H3 @ W2^T + b2    [16384x256x64] via WMMA bf16 -> fp32
    gcn_lin_wmma_kernel<HID, NCLS, false, float>
        <<<(NN / 16) * (NCLS / 16) / 8, 256, 0, stream>>>(H3, W2b, b2, out);
}
